// HydraAttention_6133213298969
// MI455X (gfx1250) — compile-verified, matrix operand order checked by reading, not run on hardware
//
#include <hip/hip_runtime.h>
#include <hip/hip_bf16.h>
#include <math.h>

typedef __attribute__((ext_vector_type(16))) __bf16 v16bf;
typedef __attribute__((ext_vector_type(8)))  float  v8f;
typedef __attribute__((ext_vector_type(4)))  __bf16 v4bf;
typedef __attribute__((__vector_size__(16))) int    v4i32;

#define BDIM  8
#define NSEQ  4096
#define DDIM  1024
#define MROWS (BDIM * NSEQ)   // 32768

#define BLK_M 128
#define BLK_N 64
#define BLK_K 32
#define LDSS  40              // padded LDS row stride (elements): 80B rows, bank-conflict free
#define A_ELEMS (BLK_M * LDSS)
#define B_ELEMS (3 * BLK_N * LDSS)
#define NKIT  (DDIM / BLK_K)

#if __has_builtin(__builtin_amdgcn_global_load_async_to_lds_b128)
#define HAS_ASYNC 1
#else
#define HAS_ASYNC 0
#endif

// 16B global -> LDS copy: async DMA path on CDNA5 (no VGPR round-trip), sync fallback
__device__ __forceinline__ void copy16(const __bf16* g, __bf16* l) {
#if HAS_ASYNC
    __builtin_amdgcn_global_load_async_to_lds_b128(
        (__attribute__((address_space(1))) v4i32*)g,
        (__attribute__((address_space(3))) v4i32*)l, 0, 0);
#else
    *(uint4*)l = *(const uint4*)g;
#endif
}

__device__ __forceinline__ void wait_async_all() {
#if HAS_ASYNC
#if __has_builtin(__builtin_amdgcn_s_wait_asynccnt)
    __builtin_amdgcn_s_wait_asynccnt(0);
#else
    asm volatile("s_wait_asynccnt 0x0" ::: "memory");
#endif
#endif
}

// ---------------- fp32 -> bf16 conversion (vectorized 4-wide) ----------------
__global__ void k_f32_to_bf16(const float* __restrict__ src, __bf16* __restrict__ dst, int n4) {
    int i = blockIdx.x * blockDim.x + threadIdx.x;
    if (i < n4) {
        float4 v = ((const float4*)src)[i];
        v4bf o;
        o[0] = (__bf16)v.x; o[1] = (__bf16)v.y; o[2] = (__bf16)v.z; o[3] = (__bf16)v.w;
        ((v4bf*)dst)[i] = o;
    }
}

// ---------------- zero-init for accumulator buffers ----------------
__global__ void k_zero(float* __restrict__ p, int n) {
    int i = blockIdx.x * blockDim.x + threadIdx.x;
    if (i < n) p[i] = 0.0f;
}

// ---------------- fused QKV GEMM: y = x @ W^T for 3 weights, + row SSQ -------
// grid: (MROWS/BLK_M, DDIM/BLK_N), block: 256 (8 wave32)
__global__ __launch_bounds__(256)
void k_gemm_qkv(const __bf16* __restrict__ xb, const __bf16* __restrict__ wb,
                float* __restrict__ qout, __bf16* __restrict__ kout, __bf16* __restrict__ vout,
                float* __restrict__ qss, float* __restrict__ kss)
{
    __shared__ __align__(16) __bf16 sA[2][A_ELEMS];   // 2 x 10240 B (double-buffered)
    __shared__ __align__(16) __bf16 sB[2][B_ELEMS];   // 2 x 15360 B

    const int tid    = threadIdx.x;
    const int lane   = tid & 31;
    const int wv     = tid >> 5;        // 0..7
    const int wave_m = wv >> 1;         // 0..3 -> 32 rows each
    const int wave_n = wv & 1;          // 0..1 -> 32 cols each
    const int lrow   = lane & 15;
    const int lh     = lane >> 4;       // lane half selects K-half (16-bit A-frag layout)

    const int m0 = blockIdx.x * BLK_M;
    const int n0 = blockIdx.y * BLK_N;

    v8f acc[3][2][2];
    #pragma unroll
    for (int w = 0; w < 3; ++w)
        #pragma unroll
        for (int tm = 0; tm < 2; ++tm)
            #pragma unroll
            for (int tn = 0; tn < 2; ++tn)
                acc[w][tm][tn] = (v8f)0.0f;

    union FragU { v16bf v; uint4 q[2]; };

    // Stage one K-slice into LDS buffer `buf`: A tile 128x32, B tiles 3x(64x32)
    auto stage = [&](int kc, int buf) {
        #pragma unroll
        for (int it = 0; it < 2; ++it) {
            int c   = tid + it * 256;
            int row = c >> 2, seg = c & 3;
            copy16(xb + (size_t)(m0 + row) * DDIM + kc + seg * 8,
                   &sA[buf][row * LDSS + seg * 8]);
        }
        int row = tid >> 2, seg = tid & 3;
        #pragma unroll
        for (int w = 0; w < 3; ++w) {
            copy16(wb + (size_t)w * DDIM * DDIM + (size_t)(n0 + row) * DDIM + kc + seg * 8,
                   &sB[buf][(w * BLK_N + row) * LDSS + seg * 8]);
        }
    };

    stage(0, 0);

    for (int i = 0; i < NKIT; ++i) {
        const int cur = i & 1;
        wait_async_all();      // my async writes to sA/sB[cur] complete
        __syncthreads();       // everyone's writes visible; prior reads of [cur^1] retired
        if (i + 1 < NKIT) stage((i + 1) * BLK_K, cur ^ 1);   // prefetch next slice

        const __bf16* cA = sA[cur];
        const __bf16* cB = sB[cur];

        // A fragments (16-bit A layout: lanes 0-15 K-halves {0-7,16-23}, lanes 16-31 {8-15,24-31})
        FragU a[2];
        #pragma unroll
        for (int tm = 0; tm < 2; ++tm) {
            int r = wave_m * 32 + tm * 16 + lrow;
            a[tm].q[0] = *(const uint4*)(cA + r * LDSS + 8 * lh);
            a[tm].q[1] = *(const uint4*)(cA + r * LDSS + 16 + 8 * lh);
        }
        #pragma unroll
        for (int w = 0; w < 3; ++w) {
            #pragma unroll
            for (int tn = 0; tn < 2; ++tn) {
                FragU b;
                int r = wave_n * 32 + tn * 16 + lrow;
                b.q[0] = *(const uint4*)(cB + (w * BLK_N + r) * LDSS + 8 * lh);
                b.q[1] = *(const uint4*)(cB + (w * BLK_N + r) * LDSS + 16 + 8 * lh);
                #pragma unroll
                for (int tm = 0; tm < 2; ++tm) {
                    acc[w][tm][tn] = __builtin_amdgcn_wmma_f32_16x16x32_bf16(
                        false, a[tm].v, false, b.v, (short)0, acc[w][tm][tn], false, false);
                }
            }
        }
    }

    // C layout: VGPR r -> row = r + 8*lh, col = lrow (within 16x16 tile)
    #pragma unroll
    for (int tm = 0; tm < 2; ++tm) {
        #pragma unroll
        for (int tn = 0; tn < 2; ++tn) {
            #pragma unroll
            for (int r = 0; r < 8; ++r) {
                int row = m0 + wave_m * 32 + tm * 16 + r + 8 * lh;
                int col = n0 + wave_n * 32 + tn * 16 + lrow;
                size_t idx = (size_t)row * DDIM + col;
                qout[idx] = acc[0][tm][tn][r];               // Q kept fp32 (final output base)
                kout[idx] = (__bf16)acc[1][tm][tn][r];
                vout[idx] = (__bf16)acc[2][tm][tn][r];
            }
        }
    }

    // Row sum-of-squares for Q and K from f32 accumulators: shuffle-reduce 16 N-lanes, one atomic per row
    #pragma unroll
    for (int w = 0; w < 2; ++w) {
        float* ss = (w == 0) ? qss : kss;
        #pragma unroll
        for (int tm = 0; tm < 2; ++tm) {
            #pragma unroll
            for (int r = 0; r < 8; ++r) {
                float v0 = acc[w][tm][0][r];
                float v1 = acc[w][tm][1][r];
                float s = v0 * v0 + v1 * v1;
                s += __shfl_xor(s, 1, 32);
                s += __shfl_xor(s, 2, 32);
                s += __shfl_xor(s, 4, 32);
                s += __shfl_xor(s, 8, 32);       // stays within each 16-lane half
                if (lrow == 0) {
                    int row = m0 + wave_m * 32 + tm * 16 + r + 8 * lh;
                    atomicAdd(&ss[row], s);
                }
            }
        }
    }
}

// ---------------- g[b,d] = (1/N) sum_n Khat * V ------------------------------
// grid: (DDIM/256, BDIM, 16 n-chunks), block 256
__global__ void k_gfeat(const __bf16* __restrict__ kout, const __bf16* __restrict__ vout,
                        const float* __restrict__ kss, float* __restrict__ gfeat)
{
    int d  = blockIdx.x * blockDim.x + threadIdx.x;
    int b  = blockIdx.y;
    int nc = blockIdx.z;
    float acc = 0.0f;
    for (int n = nc * (NSEQ / 16); n < (nc + 1) * (NSEQ / 16); ++n) {
        int row = b * NSEQ + n;
        float inv = 1.0f / fmaxf(sqrtf(kss[row]), 1e-12f);
        size_t idx = (size_t)row * DDIM + d;
        acc += ((float)kout[idx] * inv) * (float)vout[idx];
    }
    atomicAdd(&gfeat[b * DDIM + d], acc * (1.0f / (float)NSEQ));
}

// ---------------- out = Qhat * g (in place over d_out) -----------------------
__global__ void k_scale(float* __restrict__ out, const float* __restrict__ qss,
                        const float* __restrict__ gfeat)
{
    int i   = blockIdx.x * blockDim.x + threadIdx.x;
    int row = i >> 10;           // / DDIM
    int d   = i & (DDIM - 1);
    int b   = row >> 12;         // / NSEQ
    float inv = 1.0f / fmaxf(sqrtf(qss[row]), 1e-12f);
    out[i] = out[i] * inv * gfeat[b * DDIM + d];
}

extern "C" void kernel_launch(void* const* d_in, const int* in_sizes, int n_in,
                              void* d_out, int out_size, void* d_ws, size_t ws_size,
                              hipStream_t stream)
{
    const float* x  = (const float*)d_in[0];
    const float* Wq = (const float*)d_in[1];
    const float* Wk = (const float*)d_in[2];
    const float* Wv = (const float*)d_in[3];
    float* out = (float*)d_out;

    // Workspace layout
    char* ws = (char*)d_ws;
    __bf16* xb    = (__bf16*)ws;    ws += (size_t)MROWS * DDIM * 2;   // 64 MiB
    __bf16* wb    = (__bf16*)ws;    ws += (size_t)3 * DDIM * DDIM * 2;
    __bf16* kout  = (__bf16*)ws;    ws += (size_t)MROWS * DDIM * 2;
    __bf16* vout  = (__bf16*)ws;    ws += (size_t)MROWS * DDIM * 2;
    float*  qss   = (float*)ws;     ws += (size_t)MROWS * 4;
    float*  kss   = (float*)ws;     ws += (size_t)MROWS * 4;
    float*  gfeat = (float*)ws;     ws += (size_t)BDIM * DDIM * 4;

    // 1) zero accumulators (qss, kss, gfeat are contiguous)
    {
        int n = MROWS * 2 + BDIM * DDIM;
        k_zero<<<(n + 255) / 256, 256, 0, stream>>>(qss, n);
    }
    // 2) convert x and weights to bf16
    {
        int n4 = MROWS * DDIM / 4;
        k_f32_to_bf16<<<n4 / 256, 256, 0, stream>>>(x, xb, n4);
        int w4 = DDIM * DDIM / 4;
        k_f32_to_bf16<<<w4 / 256, 256, 0, stream>>>(Wq, wb + 0 * (size_t)DDIM * DDIM, w4);
        k_f32_to_bf16<<<w4 / 256, 256, 0, stream>>>(Wk, wb + 1 * (size_t)DDIM * DDIM, w4);
        k_f32_to_bf16<<<w4 / 256, 256, 0, stream>>>(Wv, wb + 2 * (size_t)DDIM * DDIM, w4);
    }
    // 3) fused QKV WMMA GEMM (async double-buffered staging) + row sum-of-squares
    {
        dim3 grid(MROWS / BLK_M, DDIM / BLK_N);
        k_gemm_qkv<<<grid, 256, 0, stream>>>(xb, wb, out, kout, vout, qss, kss);
    }
    // 4) global feature reduction over N
    {
        dim3 grid(DDIM / 256, BDIM, 16);
        k_gfeat<<<grid, 256, 0, stream>>>(kout, vout, kss, gfeat);
    }
    // 5) final scaling in place
    {
        int n = MROWS * DDIM;
        k_scale<<<n / 256, 256, 0, stream>>>(out, qss, gfeat);
    }
    (void)in_sizes; (void)n_in; (void)out_size; (void)ws_size;
}
